// EMM_11433202942123
// MI455X (gfx1250) — matrix operand order repaired and checked by reading, not compiled, not run on hardware
//
#include <hip/hip_runtime.h>

typedef __attribute__((ext_vector_type(16))) _Float16 v16h;
typedef __attribute__((ext_vector_type(2)))  __fp16   fp16x2;
typedef __attribute__((ext_vector_type(8)))  float    v8f;

#define NUMB      8          // blocks
#define CH        256        // channels (K and N)
#define M_TOTAL   16384      // 4*4096 rows per block
#define ROWSTRIDE (NUMB*CH)  // 2048 floats between consecutive rows of one block
#define NB_HALF   128        // N columns handled per workgroup
#define ROW_ELEMS 536        // combined LDS row: [0,264) w_r | [272,536) w_i  (f16)
#define WI_OFF    272        // w_i start inside combined row (544B, 16B-aligned)
#define THREADS   256        // 8 wave32
#define M_PER_WG  256        // 4 m-iters * 4 wave_m strips * 16 rows

union Frag { unsigned u[8]; v16h v; };

// One v_cvt_pk_f16_f32 per pair (RTZ), no scalar re-pack.
static __device__ __forceinline__ unsigned pkh(float a, float b) {
    union { fp16x2 h; unsigned u; } t;
    t.h = __builtin_amdgcn_cvt_pkrtz(a, b);
    return t.u;
}
static __device__ __forceinline__ unsigned short h16(float a) {
    union { _Float16 h; unsigned short s; } t;
    t.h = (_Float16)a;
    return t.s;
}
static __device__ __forceinline__ Frag loadB(const unsigned short* p) {
    Frag f;
    const uint4 t0 = *(const uint4*)p;
    const uint4 t1 = *(const uint4*)(p + 16);
    f.u[0] = t0.x; f.u[1] = t0.y; f.u[2] = t0.z; f.u[3] = t0.w;
    f.u[4] = t1.x; f.u[5] = t1.y; f.u[6] = t1.z; f.u[7] = t1.w;
    return f;
}
#define WMMA(A, B, C) __builtin_amdgcn_wmma_f32_16x16x32_f16( \
    false, (A).v, false, (B).v, (short)0, (C), false, false)

__global__ __launch_bounds__(THREADS)
void cgemm_block_wmma(const float* __restrict__ real, const float* __restrict__ imag,
                      const float* __restrict__ w_r,  const float* __restrict__ w_i,
                      const float* __restrict__ b_r,  const float* __restrict__ b_i,
                      float* __restrict__ outR, float* __restrict__ outI)
{
    extern __shared__ unsigned short smem[];   // [NB_HALF][ROW_ELEMS] interleaved wr|wi

    const int tid = threadIdx.x;
    const int blk = blockIdx.z;   // 0..7
    const int nb  = blockIdx.y;   // 0..1  (N half)
    const int mt  = blockIdx.x;   // 0..63 (M tile of 256 rows)

    // ---------------- Phase 1: stage weights f32 -> f16, transposed to [n][k] ---------
    {
        const long wbase = (long)blk * CH * CH + (long)nb * NB_HALF;
        for (int idx = tid; idx < CH * (NB_HALF / 4); idx += THREADS) {
            const int d  = idx >> 5;          // 0..255  (K index)
            const int n4 = (idx & 31) << 2;   // 0,4,...,124 (local N)
            const float4 vr = *(const float4*)(w_r + wbase + (long)d * CH + n4);
            const float4 vi = *(const float4*)(w_i + wbase + (long)d * CH + n4);
            smem[(n4 + 0) * ROW_ELEMS + d]          = h16(vr.x);
            smem[(n4 + 1) * ROW_ELEMS + d]          = h16(vr.y);
            smem[(n4 + 2) * ROW_ELEMS + d]          = h16(vr.z);
            smem[(n4 + 3) * ROW_ELEMS + d]          = h16(vr.w);
            smem[(n4 + 0) * ROW_ELEMS + WI_OFF + d] = h16(vi.x);
            smem[(n4 + 1) * ROW_ELEMS + WI_OFF + d] = h16(vi.y);
            smem[(n4 + 2) * ROW_ELEMS + WI_OFF + d] = h16(vi.z);
            smem[(n4 + 3) * ROW_ELEMS + WI_OFF + d] = h16(vi.w);
        }
    }
    __syncthreads();

    // ---------------- Phase 2: WMMA compute ------------------------------------------
    const int lane   = tid & 31;
    const int wv     = tid >> 5;     // 0..7
    const int wave_m = wv >> 1;      // 0..3
    const int wave_n = wv & 1;       // 0..1
    const int h      = lane >> 4;    // half of wave
    const int l16    = lane & 15;

    // Bias depends only on N: preload one column value per n-tile per lane.
    float biasr[4], biasi[4];
    #pragma unroll
    for (int nt = 0; nt < 4; ++nt) {
        const int ng = nb * NB_HALF + wave_n * 64 + nt * 16 + l16;
        biasr[nt] = b_r[blk * CH + ng];
        biasi[nt] = b_i[blk * CH + ng];
    }

    // One loop-invariant LDS base; all fragment addresses are base + immediate
    // (max ~52.5KB < 64KB DS offset field): no VALU address math in the hot loop,
    // hence no WMMA->VALU WAR-hazard v_nops either.
    const unsigned short* __restrict__ baseB =
        smem + (wave_n * 64 + l16) * ROW_ELEMS + h * 8;

    const v8f vzero = {0.f, 0.f, 0.f, 0.f, 0.f, 0.f, 0.f, 0.f};
    const long mwg = (long)mt * M_PER_WG;

    #pragma unroll 1
    for (int miter = 0; miter < 4; ++miter) {
        const long m0    = mwg + miter * 64 + wave_m * 16;
        const long abase = (m0 + l16) * ROWSTRIDE + (long)blk * CH;

        if (miter < 3) {  // speculative prefetch of next m-tile rows (global_prefetch_b8)
            __builtin_prefetch(real + abase + 64L * ROWSTRIDE, 0, 0);
            __builtin_prefetch(imag + abase + 64L * ROWSTRIDE, 0, 0);
        }

        v8f accr[4], acci[4];
        #pragma unroll
        for (int nt = 0; nt < 4; ++nt) { accr[nt] = vzero; acci[nt] = vzero; }

        // Unroll 2: enough pipelining without spilling (full unroll spilled to scratch).
        #pragma unroll 2
        for (int ks = 0; ks < 8; ++ks) {
            // A-fragment (16x32 f16): lane row m0+l16; K runs h*8+{0..7} and 16+h*8+{0..7}
            const int d0 = ks * 32 + h * 8;
            const float4 r0 = *(const float4*)(real + abase + d0);
            const float4 r1 = *(const float4*)(real + abase + d0 + 4);
            const float4 r2 = *(const float4*)(real + abase + d0 + 16);
            const float4 r3 = *(const float4*)(real + abase + d0 + 20);
            const float4 i0 = *(const float4*)(imag + abase + d0);
            const float4 i1 = *(const float4*)(imag + abase + d0 + 4);
            const float4 i2 = *(const float4*)(imag + abase + d0 + 16);
            const float4 i3 = *(const float4*)(imag + abase + d0 + 20);

            Frag Ar, Ai, An;
            Ar.u[0] = pkh(r0.x, r0.y); Ar.u[1] = pkh(r0.z, r0.w);
            Ar.u[2] = pkh(r1.x, r1.y); Ar.u[3] = pkh(r1.z, r1.w);
            Ar.u[4] = pkh(r2.x, r2.y); Ar.u[5] = pkh(r2.z, r2.w);
            Ar.u[6] = pkh(r3.x, r3.y); Ar.u[7] = pkh(r3.z, r3.w);
            Ai.u[0] = pkh(i0.x, i0.y); Ai.u[1] = pkh(i0.z, i0.w);
            Ai.u[2] = pkh(i1.x, i1.y); Ai.u[3] = pkh(i1.z, i1.w);
            Ai.u[4] = pkh(i2.x, i2.y); Ai.u[5] = pkh(i2.z, i2.w);
            Ai.u[6] = pkh(i3.x, i3.y); Ai.u[7] = pkh(i3.z, i3.w);
            #pragma unroll
            for (int q = 0; q < 8; ++q) An.u[q] = Ai.u[q] ^ 0x80008000u;  // -imag

            const int koff = ks * 32;   // + h*8 already folded into base pointer

            // Process n-tiles in pairs: issue all 8 ds_load_b128 for the pair up front
            // (distinct registers -> deep LDS pipelining), then 8 WMMAs.
            #pragma unroll
            for (int np = 0; np < 2; ++np) {
                const int nt0 = np * 2;
                const int nt1 = np * 2 + 1;
                const int lb0 = nt0 * 16 * ROW_ELEMS + koff;
                const int lb1 = nt1 * 16 * ROW_ELEMS + koff;

                const Frag Br0 = loadB(baseB + lb0);
                const Frag Bi0 = loadB(baseB + lb0 + WI_OFF);
                const Frag Br1 = loadB(baseB + lb1);
                const Frag Bi1 = loadB(baseB + lb1 + WI_OFF);

                // out_r += real*w_r + (-imag)*w_i ; out_i += real*w_i + imag*w_r
                accr[nt0] = WMMA(Ar, Br0, accr[nt0]);
                acci[nt0] = WMMA(Ai, Br0, acci[nt0]);
                accr[nt1] = WMMA(Ar, Br1, accr[nt1]);
                acci[nt1] = WMMA(Ai, Br1, acci[nt1]);
                accr[nt0] = WMMA(An, Bi0, accr[nt0]);
                acci[nt0] = WMMA(Ar, Bi0, acci[nt0]);
                accr[nt1] = WMMA(An, Bi1, accr[nt1]);
                acci[nt1] = WMMA(Ar, Bi1, acci[nt1]);
            }
        }

        // Bias + store. C/D layout: VGPR j -> M = j + 8*h, N = l16 (within 16x16 tile).
        #pragma unroll
        for (int nt = 0; nt < 4; ++nt) {
            const long ng = nb * NB_HALF + wave_n * 64 + nt * 16 + l16;
            #pragma unroll
            for (int j = 0; j < 8; ++j) {
                const long row = m0 + j + 8 * h;
                const long off = row * ROWSTRIDE + (long)blk * CH + ng;
                outR[off] = accr[nt][j] + biasr[nt];
                outI[off] = acci[nt][j] + biasi[nt];
            }
        }
    }
}

extern "C" void kernel_launch(void* const* d_in, const int* in_sizes, int n_in,
                              void* d_out, int out_size, void* d_ws, size_t ws_size,
                              hipStream_t stream) {
    const float* real = (const float*)d_in[0];
    const float* imag = (const float*)d_in[1];
    const float* w_r  = (const float*)d_in[2];
    const float* w_i  = (const float*)d_in[3];
    const float* b_r  = (const float*)d_in[4];
    const float* b_i  = (const float*)d_in[5];
    float* outR = (float*)d_out;
    float* outI = outR + (long)M_TOTAL * NUMB * CH;   // out tuple concatenated flat

    const dim3 grid(M_TOTAL / M_PER_WG, CH / NB_HALF, NUMB);  // (64, 2, 8)
    const dim3 block(THREADS);
    const size_t smem_bytes = (size_t)NB_HALF * ROW_ELEMS * sizeof(unsigned short); // 137216

    cgemm_block_wmma<<<grid, block, smem_bytes, stream>>>(
        real, imag, w_r, w_i, b_r, b_i, outR, outI);
}
